// VertexOnlyMPNN_62680752718357
// MI455X (gfx1250) — matrix operand order repaired and checked by reading, not compiled, not run on hardware
//
#include <hip/hip_runtime.h>
#include <hip/hip_bf16.h>
#include <math.h>

// ---------------------------------------------------------------------------
// CDNA5 (gfx1250) MPNN: bf16 WMMA 16x16x32, f32 accumulate.
// Exploits h=0 at iter 0:  m0(v) = indeg(v)*relu(b_msg),
//                          h1 = tanh(x @ W_upd_x + indeg*u + b_upd),
//                          u  = relu(b_msg) @ W_upd_m.
// Heavy iter-1 edge GEMM + scatter done with gathered-A WMMA tiles.
// Full tiles (the common case: N,E divisible by 16) take an unpredicated
// fast path so the hot loops carry no exec-mask cascades.
// ---------------------------------------------------------------------------

typedef __bf16 bf16_t;
typedef __attribute__((ext_vector_type(16))) __bf16 v16bf;
typedef __attribute__((ext_vector_type(8)))  float  v8f;

union Frag { v16bf v; unsigned u[8]; };

__device__ __forceinline__ unsigned pack2(float a, float b) {
    union { bf16_t h[2]; unsigned u; } x;
    x.h[0] = (bf16_t)a; x.h[1] = (bf16_t)b; return x.u;
}

// A-fragment (16x32 bf16, MxK) from a row-major bf16 row pointer.
// ISA layout: lanes 0-15 row r hold K kb+[0..7] (v0..3) and kb+[16..23] (v4..7);
// lanes 16-31 hold kb+[8..15] and kb+[24..31].
__device__ __forceinline__ void load_a_bf16(Frag& f, const bf16_t* rowp, int kb, int half) {
    const unsigned* p = (const unsigned*)(rowp + kb + half * 8);
    const unsigned* q = (const unsigned*)(rowp + kb + half * 8 + 16);
#pragma unroll
    for (int j = 0; j < 4; ++j) { f.u[j] = p[j]; f.u[4 + j] = q[j]; }
}

// Same, but source row is f32: convert to bf16 while packing.
__device__ __forceinline__ void load_a_f32(Frag& f, const float* rowp, int kb, int half) {
    const float* p = rowp + kb + half * 8;
    const float* q = p + 16;
#pragma unroll
    for (int j = 0; j < 4; ++j) {
        f.u[j]     = pack2(p[2 * j], p[2 * j + 1]);
        f.u[4 + j] = pack2(q[2 * j], q[2 * j + 1]);
    }
}

// B-fragment (32x16 bf16, KxN tile t of a Kx128 matrix pair-packed in LDS).
// lanes 0-15 col n hold K 32c+[0..7],[16..23]; lanes 16-31 hold 32c+[8..15],[24..31].
__device__ __forceinline__ void load_b_lds(Frag& f, const unsigned* ldsB, int c, int t, int lane) {
    int col = (lane & 15) + (t << 4);
    int kp0 = (c << 4) + ((lane >> 4) << 2);
#pragma unroll
    for (int j = 0; j < 8; ++j) f.u[j] = ldsB[(kp0 + j + (j >= 4 ? 4 : 0)) * 128 + col];
}

__device__ __forceinline__ v8f wmma_bf16(const Frag& a, const Frag& b, v8f c) {
    return __builtin_amdgcn_wmma_f32_16x16x32_bf16(false, a.v, false, b.v,
                                                   (short)0, c, false, false);
}

// Convert a Kx128 f32 weight matrix to pair-packed bf16 in LDS.
__device__ __forceinline__ void pack_W_lds(unsigned* ldsB, const float* W, int K,
                                           int tid, int nthr) {
    int total = (K >> 1) * 128;
    for (int i = tid; i < total; i += nthr) {
        int kp = i >> 7, col = i & 127;
        ldsB[i] = pack2(W[(2 * kp) * 128 + col], W[(2 * kp + 1) * 128 + col]);
    }
}

// ------------------------------- small kernels ------------------------------

__global__ void deg_kernel(const int* __restrict__ edst, int* __restrict__ deg, int E) {
    int i = blockIdx.x * blockDim.x + threadIdx.x;
    if (i < E) atomicAdd(&deg[edst[i]], 1);
}

// u[j] = sum_k relu(b_msg[k]) * W_upd[k,j]   (W_upd rows 0..127 = message part)
__global__ void u_kernel(const float* __restrict__ b_msg, const float* __restrict__ W_upd,
                         float* __restrict__ u) {
    int j = threadIdx.x;
    float s = 0.f;
    for (int k = 0; k < 128; ++k) {
        float r = b_msg[k]; r = r > 0.f ? r : 0.f;
        s += r * W_upd[k * 128 + j];
    }
    u[j] = s;
}

// m[v,j] = deg[v] * relu(b_msg[j])
__global__ void minit_kernel(const int* __restrict__ deg, const float* __restrict__ b_msg,
                             float* __restrict__ m, int total) {
    int i = blockIdx.x * blockDim.x + threadIdx.x;
    if (i < total) {
        int v = i >> 7, j = i & 127;
        float r = b_msg[j]; r = r > 0.f ? r : 0.f;
        m[i] = (float)deg[v] * r;
    }
}

// out[j] = relu(sum_k hsum[k] * W_ro[k,j] + b_ro[j])
__global__ void ro_kernel(const float* __restrict__ hsum, const float* __restrict__ W_ro,
                          const float* __restrict__ b_ro, float* __restrict__ out) {
    int j = threadIdx.x;
    float s = b_ro[j];
    for (int k = 0; k < 128; ++k) s += hsum[k] * W_ro[k * 128 + j];
    out[j] = s > 0.f ? s : 0.f;
}

// ------------------------------- WMMA kernels -------------------------------

// x_bf = bf16(data @ W_emb + b_emb)            [N,64] x [64,128]
__global__ __launch_bounds__(256) void emb_kernel(const float* __restrict__ data,
                                                  const float* __restrict__ W_emb,
                                                  const float* __restrict__ b_emb,
                                                  bf16_t* __restrict__ x_bf, int N) {
    __shared__ unsigned ldsB[32 * 128];  // K=64 -> 32 pair rows, 16 KB
    pack_W_lds(ldsB, W_emb, 64, threadIdx.x, 256);
    __syncthreads();
    int lane = threadIdx.x & 31, wave = threadIdx.x >> 5;
    int v0 = (blockIdx.x * 8 + wave) * 16;
    if (v0 >= N) return;
    const bool full = (v0 + 16 <= N);            // wave-uniform
    int half = lane >> 4;
    int row = v0 + (lane & 15); if (row >= N) row = N - 1;

    Frag a[2];
#pragma unroll
    for (int c = 0; c < 2; ++c) load_a_f32(a[c], data + (size_t)row * 64, 32 * c, half);

    for (int t = 0; t < 8; ++t) {
        int col = (lane & 15) + 16 * t;
        float bias = b_emb[col];
        v8f acc;
#pragma unroll
        for (int i = 0; i < 8; ++i) acc[i] = bias;
        Frag b;
#pragma unroll
        for (int c = 0; c < 2; ++c) { load_b_lds(b, ldsB, c, t, lane); acc = wmma_bf16(a[c], b, acc); }
        bf16_t* outp = x_bf + (size_t)(v0 + 8 * half) * 128 + col;
        if (full) {
#pragma unroll
            for (int i = 0; i < 8; ++i) outp[(size_t)i * 128] = (bf16_t)acc[i];
        } else {
#pragma unroll
            for (int i = 0; i < 8; ++i)
                if (v0 + i + 8 * half < N) outp[(size_t)i * 128] = (bf16_t)acc[i];
        }
    }
}

// h1 = bf16(tanh(x @ W_upd[256:384] + deg*u + b_upd))     [N,128] x [128,128]
__global__ __launch_bounds__(256) void h1_kernel(const bf16_t* __restrict__ x_bf,
                                                 const float* __restrict__ W_upd,
                                                 const float* __restrict__ b_upd,
                                                 const float* __restrict__ u,
                                                 const int* __restrict__ deg,
                                                 bf16_t* __restrict__ h1, int N) {
    __shared__ unsigned ldsB[64 * 128];  // K=128 (x-part), 32 KB
    {
        int total = 64 * 128;
        for (int i = threadIdx.x; i < total; i += 256) {
            int kp = i >> 7, col = i & 127;
            ldsB[i] = pack2(W_upd[(256 + 2 * kp) * 128 + col],
                            W_upd[(256 + 2 * kp + 1) * 128 + col]);
        }
    }
    __syncthreads();
    int lane = threadIdx.x & 31, wave = threadIdx.x >> 5;
    int v0 = (blockIdx.x * 8 + wave) * 16;
    if (v0 >= N) return;
    const bool full = (v0 + 16 <= N);
    int half = lane >> 4;
    int row = v0 + (lane & 15); if (row >= N) row = N - 1;

    Frag a[4];
#pragma unroll
    for (int c = 0; c < 4; ++c) load_a_bf16(a[c], x_bf + (size_t)row * 128, 32 * c, half);

    float dg[8];
#pragma unroll
    for (int i = 0; i < 8; ++i) {
        int rr = v0 + i + 8 * half; if (rr >= N) rr = N - 1;
        dg[i] = (float)deg[rr];
    }

    for (int t = 0; t < 8; ++t) {
        int col = (lane & 15) + 16 * t;
        float bu = b_upd[col], uu = u[col];
        v8f acc;
#pragma unroll
        for (int i = 0; i < 8; ++i) acc[i] = bu + dg[i] * uu;
        Frag b;
#pragma unroll
        for (int c = 0; c < 4; ++c) { load_b_lds(b, ldsB, c, t, lane); acc = wmma_bf16(a[c], b, acc); }
        bf16_t* outp = h1 + (size_t)(v0 + 8 * half) * 128 + col;
        if (full) {
#pragma unroll
            for (int i = 0; i < 8; ++i) outp[(size_t)i * 128] = (bf16_t)tanhf(acc[i]);
        } else {
#pragma unroll
            for (int i = 0; i < 8; ++i)
                if (v0 + i + 8 * half < N) outp[(size_t)i * 128] = (bf16_t)tanhf(acc[i]);
        }
    }
}

// msg = relu(concat(h1[dst], h1[src]) @ W_msg + b_msg); m[dst] += msg  (atomic)
__global__ __launch_bounds__(256) void msg_kernel(const bf16_t* __restrict__ h1,
                                                  const int* __restrict__ esrc,
                                                  const int* __restrict__ edst,
                                                  const float* __restrict__ W_msg,
                                                  const float* __restrict__ b_msg,
                                                  float* __restrict__ m, int E) {
    extern __shared__ unsigned ldsB[];   // 128*128 u32 = 64 KB (K=256 pair-packed)
    pack_W_lds(ldsB, W_msg, 256, threadIdx.x, 256);
    __syncthreads();
    int lane = threadIdx.x & 31, wave = threadIdx.x >> 5;
    int e0 = (blockIdx.x * 8 + wave) * 16;
    if (e0 >= E) return;
    const bool full = (e0 + 16 <= E);
    int half = lane >> 4;
    int e_r = e0 + (lane & 15); if (e_r >= E) e_r = E - 1;
    int d_r = edst[e_r], s_r = esrc[e_r];

    Frag a[8];   // K = 256: chunks 0-3 from h1[dst], 4-7 from h1[src]
#pragma unroll
    for (int c = 0; c < 4; ++c) load_a_bf16(a[c],     h1 + (size_t)d_r * 128, 32 * c, half);
#pragma unroll
    for (int c = 0; c < 4; ++c) load_a_bf16(a[4 + c], h1 + (size_t)s_r * 128, 32 * c, half);

    int nd[8];
#pragma unroll
    for (int i = 0; i < 8; ++i) {
        int ei = e0 + i + 8 * half;
        nd[i] = edst[ei < E ? ei : (E - 1)];
    }

    for (int t = 0; t < 8; ++t) {
        int col = (lane & 15) + 16 * t;
        float bias = b_msg[col];
        v8f acc;
#pragma unroll
        for (int i = 0; i < 8; ++i) acc[i] = bias;
        Frag b;
#pragma unroll
        for (int c = 0; c < 8; ++c) { load_b_lds(b, ldsB, c, t, lane); acc = wmma_bf16(a[c], b, acc); }
        if (full) {
#pragma unroll
            for (int i = 0; i < 8; ++i) {
                float v = acc[i] > 0.f ? acc[i] : 0.f;
                atomicAdd(&m[(size_t)nd[i] * 128 + col], v);
            }
        } else {
#pragma unroll
            for (int i = 0; i < 8; ++i) {
                if (e0 + i + 8 * half < E) {
                    float v = acc[i] > 0.f ? acc[i] : 0.f;
                    atomicAdd(&m[(size_t)nd[i] * 128 + col], v);
                }
            }
        }
    }
}

// hsum += column-sums of tanh(concat(m, h1, x) @ W_upd + b_upd)   [N,384]x[384,128]
__global__ __launch_bounds__(256) void upd_kernel(const float* __restrict__ m,
                                                  const bf16_t* __restrict__ h1,
                                                  const bf16_t* __restrict__ x_bf,
                                                  const float* __restrict__ W_upd,
                                                  const float* __restrict__ b_upd,
                                                  float* __restrict__ hsum, int N) {
    extern __shared__ unsigned ldsB[];   // 192*128 u32 = 96 KB (K=384 pair-packed)
    pack_W_lds(ldsB, W_upd, 384, threadIdx.x, 256);
    __syncthreads();
    int lane = threadIdx.x & 31, wave = threadIdx.x >> 5;
    int v0 = (blockIdx.x * 8 + wave) * 16;
    if (v0 >= N) return;
    const bool full = (v0 + 16 <= N);
    int half = lane >> 4;
    int row = v0 + (lane & 15); if (row >= N) row = N - 1;

    Frag a[12];  // chunks 0-3: m (f32->bf16), 4-7: h1, 8-11: x
#pragma unroll
    for (int c = 0; c < 4; ++c) load_a_f32 (a[c],     m    + (size_t)row * 128, 32 * c, half);
#pragma unroll
    for (int c = 0; c < 4; ++c) load_a_bf16(a[4 + c], h1   + (size_t)row * 128, 32 * c, half);
#pragma unroll
    for (int c = 0; c < 4; ++c) load_a_bf16(a[8 + c], x_bf + (size_t)row * 128, 32 * c, half);

    for (int t = 0; t < 8; ++t) {
        int col = (lane & 15) + 16 * t;
        float bu = b_upd[col];
        v8f acc;
#pragma unroll
        for (int i = 0; i < 8; ++i) acc[i] = bu;
        Frag b;
#pragma unroll
        for (int c = 0; c < 12; ++c) { load_b_lds(b, ldsB, c, t, lane); acc = wmma_bf16(a[c], b, acc); }
        float s = 0.f;
        if (full) {
#pragma unroll
            for (int i = 0; i < 8; ++i) s += tanhf(acc[i]);
        } else {
#pragma unroll
            for (int i = 0; i < 8; ++i)
                if (v0 + i + 8 * half < N) s += tanhf(acc[i]);
        }
        atomicAdd(&hsum[col], s);
    }
}

// ------------------------------- launch -------------------------------------

extern "C" void kernel_launch(void* const* d_in, const int* in_sizes, int n_in,
                              void* d_out, int out_size, void* d_ws, size_t ws_size,
                              hipStream_t stream) {
    const float* data  = (const float*)d_in[0];
    const int*   esrc  = (const int*)  d_in[1];
    const int*   edst  = (const int*)  d_in[2];
    const float* W_emb = (const float*)d_in[3];
    const float* b_emb = (const float*)d_in[4];
    const float* W_msg = (const float*)d_in[5];
    const float* b_msg = (const float*)d_in[6];
    const float* W_upd = (const float*)d_in[7];
    const float* b_upd = (const float*)d_in[8];
    const float* W_ro  = (const float*)d_in[9];
    const float* b_ro  = (const float*)d_in[10];
    float* out = (float*)d_out;

    const int N = in_sizes[0] / 64;     // 20000
    const int E = in_sizes[1];          // 320000

    auto align256 = [](size_t x) { return (x + 255) & ~(size_t)255; };
    char* ws = (char*)d_ws;
    size_t off = 0;
    bf16_t* x_bf = (bf16_t*)(ws + off); off = align256(off + (size_t)N * 128 * sizeof(bf16_t));
    bf16_t* h1   = (bf16_t*)(ws + off); off = align256(off + (size_t)N * 128 * sizeof(bf16_t));
    float*  m    = (float*) (ws + off); off = align256(off + (size_t)N * 128 * sizeof(float));
    int*    deg  = (int*)   (ws + off); off = align256(off + (size_t)N * sizeof(int));
    float*  u    = (float*) (ws + off); off = align256(off + 128 * sizeof(float));
    float*  hsum = (float*) (ws + off); off = align256(off + 128 * sizeof(float));

    const int tilesN = (N + 15) / 16, blocksN = (tilesN + 7) / 8;
    const int tilesE = (E + 15) / 16, blocksE = (tilesE + 7) / 8;

    // degree + analytic iter-0
    hipMemsetAsync(deg, 0, (size_t)N * sizeof(int), stream);
    deg_kernel<<<(E + 255) / 256, 256, 0, stream>>>(edst, deg, E);
    u_kernel<<<1, 128, 0, stream>>>(b_msg, W_upd, u);
    minit_kernel<<<(N * 128 + 255) / 256, 256, 0, stream>>>(deg, b_msg, m, N * 128);

    // x = data@W_emb+b_emb ; h1 = tanh(x@W_upd_x + deg*u + b_upd)
    emb_kernel<<<blocksN, 256, 0, stream>>>(data, W_emb, b_emb, x_bf, N);
    h1_kernel<<<blocksN, 256, 0, stream>>>(x_bf, W_upd, b_upd, u, deg, h1, N);

    // iteration 1: edge messages + scatter, then update reduced into hsum
    msg_kernel<<<blocksE, 256, 128 * 128 * sizeof(unsigned), stream>>>(
        h1, esrc, edst, W_msg, b_msg, m, E);
    hipMemsetAsync(hsum, 0, 128 * sizeof(float), stream);
    upd_kernel<<<blocksN, 256, 192 * 128 * sizeof(unsigned), stream>>>(
        m, h1, x_bf, W_upd, b_upd, hsum, N);

    // readout
    ro_kernel<<<1, 128, 0, stream>>>(hsum, W_ro, b_ro, out);
}
